// GraphFallbackSolver_63118839382258
// MI455X (gfx1250) — matrix operand order repaired and checked by reading, not compiled, not run on hardware
//
#include <hip/hip_runtime.h>
#include <hip/hip_bf16.h>
#include <stdint.h>

// ---------------- problem constants ----------------
#define BB     2
#define NN     10000
#define BN     20000          // BB*NN
#define C_IN   3
#define C_HID  64
#define E_PAD  262144
#define K_IN   70             // 3 + 3 + 64
#define K_INP  96             // padded to multiple of 32
#define K1     512
#define K2     256

// edge kernel tiling
#define EBLOCK 256            // 8 wave32
#define MTILE  32             // edges per workgroup

// workspace layout (bytes)
#define FYH_OFF  ((size_t)0)                       // BN*64 f16           = 2,560,000
#define WK1_OFF  ((size_t)2560000)                 // 96*512 f16 swizzled =    98,304
#define WK2_OFF  ((size_t)2658304)                 // 512*256 f16         =   262,144
#define WK3_OFF  ((size_t)2920448)                 // 256*64 f16          =    32,768
#define AGG_OFF  ((size_t)2953216)                 // (BN+1)*64 f32       = 5,120,256
#define CNT_OFF  ((size_t)8073472)                 // (BN+1)   f32        =    80,004

typedef __attribute__((ext_vector_type(16))) _Float16 v16h;
typedef __attribute__((ext_vector_type(8)))  float    v8f;

// hardware f32 global atomic add (global_atomic_add_f32), not a CAS loop
__device__ __forceinline__ void atomic_add_f32(float* p, float v) {
  unsafeAtomicAdd(p, v);
}

// Branch-free GELU (tanh form) from raw v_exp_f32 / v_rcp_f32.
// tanh(u) = 1 - 2/(exp2(u*2*log2e) + 1); saturates correctly for |u| large.
__device__ __forceinline__ float gelu_fast(float x) {
  const float u = x * (0.7978845608028654f + 0.03567740813636141f * x * x);
  const float e = __builtin_amdgcn_exp2f(u * 2.8853900817779268f);  // exp(2u)
  const float t = 1.0f - 2.0f * __builtin_amdgcn_rcpf(e + 1.0f);    // tanh(u)
  return 0.5f * x * (1.0f + t);
}

// exact-erf gelu for the tiny node/output MLPs (negligible cost there)
__device__ __forceinline__ float gelu_exact(float x) {
  return 0.5f * x * (1.0f + erff(x * 0.70710678118654752f));
}

// ------------- fragment loaders (CDNA5 wave32 WMMA layouts) -------------
// A (16x32 f16): lane L -> row m=L%16, h=L/16. element e: K = e + 8h (+8 if e>=8).
// Row-major LDS source, stride32 = stride_in_halves/2, kb32 = kbase/2.
__device__ __forceinline__ v16h ld_afrag(const _Float16* __restrict__ row0,
                                         int stride32, int kb32, int lane) {
  const int m = lane & 15, h = lane >> 4;
  const uint32_t* rp = (const uint32_t*)row0 + (size_t)m * stride32 + kb32 + 4 * h;
  union { v16h v; uint4 q[2]; } r;
  r.q[0] = *(const uint4*)(rp);
  r.q[1] = *(const uint4*)(rp + 8);
  return r.v;
}

// B (32x16 f16): pre-swizzled in global: tile = nt*KT+kt, per-lane 16 contiguous
// halves (col n = L%16, K = 8*(L/16) + (e&7) + 16*(e/8)).
__device__ __forceinline__ v16h ld_bfrag(const _Float16* __restrict__ tileBase, int lane) {
  const uint4* p = (const uint4*)(tileBase + (size_t)lane * 16);
  union { v16h v; uint4 q[2]; } r;
  r.q[0] = p[0];
  r.q[1] = p[1];
  return r.v;
}

#define WMMA_F16(A, B, C) \
  __builtin_amdgcn_wmma_f32_16x16x32_f16(false, (A), false, (B), (short)0, (C), false, false)

// ---------------- kernel: zero accumulators ----------------
__global__ void zero_kernel(float* __restrict__ agg, float* __restrict__ cnt,
                            size_t na, size_t nc) {
  size_t i = (size_t)blockIdx.x * blockDim.x + threadIdx.x;
  size_t st = (size_t)gridDim.x * blockDim.x;
  for (size_t j = i; j < na; j += st) agg[j] = 0.0f;
  for (size_t j = i; j < nc; j += st) cnt[j] = 0.0f;
}

// ---------------- kernel: node MLP  f_y = gelu(x@Wl1+b)@Wl2+b  -> f16 ----------------
__global__ __launch_bounds__(64) void node_mlp_kernel(
    const float* __restrict__ feats, const float* __restrict__ Wl1,
    const float* __restrict__ bl1, const float* __restrict__ Wl2,
    const float* __restrict__ bl2, _Float16* __restrict__ fyh) {
  __shared__ float h1[C_HID];
  const int node = blockIdx.x, t = threadIdx.x;
  float x = bl1[t];
#pragma unroll
  for (int i = 0; i < C_IN; ++i) x += feats[(size_t)node * C_IN + i] * Wl1[i * C_HID + t];
  h1[t] = gelu_exact(x);
  __syncthreads();
  float y = bl2[t];
#pragma unroll 8
  for (int j = 0; j < C_HID; ++j) y += h1[j] * Wl2[j * C_HID + t];
  fyh[(size_t)node * C_HID + t] = (_Float16)y;
}

// ---------------- kernel: convert+swizzle weights into B-fragment order ----------------
// out[(nt*KT + kt)*512 + L*16 + e] = W[kt*32 + 8*(L/16) + (e&7) + 16*(e/8)][nt*16 + L%16]  (0 if k>=Kact)
__global__ void prep_w_kernel(const float* __restrict__ W, _Float16* __restrict__ out,
                              int Kact, int Kpad, int Nn) {
  const int KT = Kpad >> 5;
  const size_t total = (size_t)KT * (Nn >> 4) * 512;
  for (size_t i = (size_t)blockIdx.x * blockDim.x + threadIdx.x; i < total;
       i += (size_t)gridDim.x * blockDim.x) {
    const int e = (int)(i & 15);
    const int L = (int)((i >> 4) & 31);
    const size_t tile = i >> 9;
    const int kt = (int)(tile % KT), nt = (int)(tile / KT);
    const int n = nt * 16 + (L & 15);
    const int k = kt * 32 + 8 * (L >> 4) + (e & 7) + 16 * (e >> 3);
    const float v = (k < Kact) ? W[(size_t)k * Nn + n] : 0.0f;
    out[i] = (_Float16)v;
  }
}

// ---------------- kernel: edge MLP (WMMA) + message scatter ----------------
__global__ __launch_bounds__(EBLOCK) void edge_mlp_kernel(
    const float* __restrict__ pts, const _Float16* __restrict__ fyh,
    const int* __restrict__ esrc, const int* __restrict__ edst,
    const _Float16* __restrict__ wk1, const float* __restrict__ bk1,
    const _Float16* __restrict__ wk2, const float* __restrict__ bk2,
    const _Float16* __restrict__ wk3, const float* __restrict__ bk3,
    float* __restrict__ agg, float* __restrict__ cnt) {
  __shared__ alignas(16) _Float16 sh_ein[MTILE * K_INP];   //  6 KB
  __shared__ alignas(16) _Float16 sh_a1[MTILE * K1];       // 32 KB
  __shared__ alignas(16) _Float16 sh_a2[MTILE * K2];       // 16 KB

  const int tid = threadIdx.x;
  const int lane = tid & 31, wave = tid >> 5;
  const int e0 = blockIdx.x * MTILE;

  // ---- gather e_in = [pts[src](3) | pts[dst](3) | f_y[src](64) | pad->96], f16 ----
  for (int idx = tid; idx < MTILE * K_INP; idx += EBLOCK) {
    const int m = idx / K_INP, c = idx - m * K_INP;
    const int e = e0 + m;
    float v = 0.0f;
    if (c < 3) {
      v = pts[(size_t)esrc[e] * 3 + c];
    } else if (c < 6) {
      v = pts[(size_t)edst[e] * 3 + (c - 3)];
    } else if (c < K_IN) {
      v = (float)fyh[(size_t)esrc[e] * C_HID + (c - 6)];
    }
    sh_ein[idx] = (_Float16)v;
  }
  if (tid < MTILE) atomic_add_f32(&cnt[edst[e0 + tid]], 1.0f);
  __syncthreads();

  const int h = lane >> 4;
  const int nl = lane & 15;

  // ---- layer 1: [32x96] x [96x512] -> gelu -> sh_a1 ----
  // per wave: 2 mtiles x 4 ntiles (ntiles wave*4 .. wave*4+3)
  {
    const int nt0 = wave * 4;
    v8f acc[2][4];
#pragma unroll
    for (int j = 0; j < 4; ++j) {
      const float bv = bk1[(nt0 + j) * 16 + nl];
      acc[0][j] = (v8f){bv, bv, bv, bv, bv, bv, bv, bv};
      acc[1][j] = acc[0][j];
    }
#pragma unroll
    for (int kc = 0; kc < K_INP / 32; ++kc) {
      const v16h a0 = ld_afrag(sh_ein, K_INP / 2, kc * 16, lane);
      const v16h a1 = ld_afrag(sh_ein + 16 * K_INP, K_INP / 2, kc * 16, lane);
#pragma unroll
      for (int j = 0; j < 4; ++j) {
        const v16h b = ld_bfrag(wk1 + ((size_t)((nt0 + j) * (K_INP / 32) + kc)) * 512, lane);
        acc[0][j] = WMMA_F16(a0, b, acc[0][j]);
        acc[1][j] = WMMA_F16(a1, b, acc[1][j]);
      }
    }
#pragma unroll
    for (int mt = 0; mt < 2; ++mt)
#pragma unroll
      for (int j = 0; j < 4; ++j) {
        const int n = (nt0 + j) * 16 + nl;
#pragma unroll
        for (int r = 0; r < 8; ++r) {
          const int m = mt * 16 + r + 8 * h;
          sh_a1[(size_t)m * K1 + n] = (_Float16)gelu_fast(acc[mt][j][r]);
        }
      }
  }
  __syncthreads();

  // ---- layer 2: [32x512] x [512x256] -> gelu -> sh_a2 ----
  // per wave: 2 mtiles x 2 ntiles (ntiles wave*2, wave*2+1)
  {
    const int nt0 = wave * 2;
    v8f acc[2][2];
#pragma unroll
    for (int j = 0; j < 2; ++j) {
      const float bv = bk2[(nt0 + j) * 16 + nl];
      acc[0][j] = (v8f){bv, bv, bv, bv, bv, bv, bv, bv};
      acc[1][j] = acc[0][j];
    }
#pragma unroll 4
    for (int kc = 0; kc < K1 / 32; ++kc) {
      const v16h a0 = ld_afrag(sh_a1, K1 / 2, kc * 16, lane);
      const v16h a1 = ld_afrag(sh_a1 + 16 * K1, K1 / 2, kc * 16, lane);
#pragma unroll
      for (int j = 0; j < 2; ++j) {
        const v16h b = ld_bfrag(wk2 + ((size_t)((nt0 + j) * (K1 / 32) + kc)) * 512, lane);
        acc[0][j] = WMMA_F16(a0, b, acc[0][j]);
        acc[1][j] = WMMA_F16(a1, b, acc[1][j]);
      }
    }
#pragma unroll
    for (int mt = 0; mt < 2; ++mt)
#pragma unroll
      for (int j = 0; j < 2; ++j) {
        const int n = (nt0 + j) * 16 + nl;
#pragma unroll
        for (int r = 0; r < 8; ++r) {
          const int m = mt * 16 + r + 8 * h;
          sh_a2[(size_t)m * K2 + n] = (_Float16)gelu_fast(acc[mt][j][r]);
        }
      }
  }
  __syncthreads();

  // ---- layer 3: [32x256] x [256x64] (+bias), msg = k * f_y[src], scatter-add ----
  {
    const int t = wave;  // exactly 8 tiles: 2 mtiles x 4 ntiles
    const int mtile = t & 1, ntile = t >> 1;
    const int n = ntile * 16 + nl;  // 0..63
    const float bv = bk3[n];
    v8f acc = {bv, bv, bv, bv, bv, bv, bv, bv};
    const _Float16* arow = sh_a2 + (size_t)mtile * 16 * K2;
#pragma unroll
    for (int kc = 0; kc < K2 / 32; ++kc) {
      const v16h a = ld_afrag(arow, K2 / 2, kc * 16, lane);
      const v16h b = ld_bfrag(wk3 + ((size_t)(ntile * (K2 / 32) + kc)) * 512, lane);
      acc = WMMA_F16(a, b, acc);
    }
    // This lane's 8 results are 8 *consecutive* edges (m = mtile*16 + 8h + r' order
    // below). edge_dst is globally non-decreasing, so combine runs of equal dst in
    // registers before issuing hardware f32 atomics.
    float pend = 0.0f;
    int dprev = -1;
#pragma unroll
    for (int r = 0; r < 8; ++r) {
      const int mloc = mtile * 16 + r + 8 * h;
      const int e = e0 + mloc;
      const float fy = (float)sh_ein[(size_t)mloc * K_INP + 6 + n];
      const float val = acc[r] * fy;
      const int d = edst[e];
      if (d != dprev) {
        if (dprev >= 0) atomic_add_f32(&agg[(size_t)dprev * C_HID + n], pend);
        dprev = d;
        pend = val;
      } else {
        pend += val;
      }
    }
    if (dprev >= 0) atomic_add_f32(&agg[(size_t)dprev * C_HID + n], pend);
  }
}

// ---------------- kernel: h = agg/max(cnt,1); out = gelu(h@Wp1+b)@Wp2+b ----------------
__global__ __launch_bounds__(64) void final_mlp_kernel(
    const float* __restrict__ agg, const float* __restrict__ cnt,
    const float* __restrict__ Wp1, const float* __restrict__ bp1,
    const float* __restrict__ Wp2, const float* __restrict__ bp2,
    float* __restrict__ out) {
  __shared__ float hv[C_HID];
  __shared__ float p1[C_HID];
  const int node = blockIdx.x, t = threadIdx.x;
  float c = cnt[node];
  c = (c < 1.0f) ? 1.0f : c;
  hv[t] = agg[(size_t)node * C_HID + t] / c;
  __syncthreads();
  float x = bp1[t];
#pragma unroll 8
  for (int j = 0; j < C_HID; ++j) x += hv[j] * Wp1[j * C_HID + t];
  p1[t] = gelu_exact(x) * Wp2[t];
  __syncthreads();
  if (t == 0) {
    float s = bp2[0];
#pragma unroll 8
    for (int j = 0; j < C_HID; ++j) s += p1[j];
    out[node] = s;
  }
}

// ---------------- launch ----------------
extern "C" void kernel_launch(void* const* d_in, const int* in_sizes, int n_in,
                              void* d_out, int out_size, void* d_ws, size_t ws_size,
                              hipStream_t stream) {
  const float* pts   = (const float*)d_in[0];
  const float* feats = (const float*)d_in[1];
  const int*   esrc  = (const int*)d_in[2];
  const int*   edst  = (const int*)d_in[3];
  const float* Wl1 = (const float*)d_in[4];  const float* bl1 = (const float*)d_in[5];
  const float* Wl2 = (const float*)d_in[6];  const float* bl2 = (const float*)d_in[7];
  const float* Wk1 = (const float*)d_in[8];  const float* bk1 = (const float*)d_in[9];
  const float* Wk2 = (const float*)d_in[10]; const float* bk2 = (const float*)d_in[11];
  const float* Wk3 = (const float*)d_in[12]; const float* bk3 = (const float*)d_in[13];
  const float* Wp1 = (const float*)d_in[14]; const float* bp1 = (const float*)d_in[15];
  const float* Wp2 = (const float*)d_in[16]; const float* bp2 = (const float*)d_in[17];

  char* ws = (char*)d_ws;
  _Float16* fyh  = (_Float16*)(ws + FYH_OFF);
  _Float16* wk1s = (_Float16*)(ws + WK1_OFF);
  _Float16* wk2s = (_Float16*)(ws + WK2_OFF);
  _Float16* wk3s = (_Float16*)(ws + WK3_OFF);
  float*    agg  = (float*)(ws + AGG_OFF);
  float*    cnt  = (float*)(ws + CNT_OFF);

  zero_kernel<<<512, 256, 0, stream>>>(agg, cnt, (size_t)(BN + 1) * C_HID, (size_t)(BN + 1));
  node_mlp_kernel<<<BN, 64, 0, stream>>>(feats, Wl1, bl1, Wl2, bl2, fyh);
  prep_w_kernel<<<96, 256, 0, stream>>>(Wk1, wk1s, K_IN, K_INP, K1);   // 70->96 x 512
  prep_w_kernel<<<256, 256, 0, stream>>>(Wk2, wk2s, K1, K1, K2);       // 512 x 256
  prep_w_kernel<<<64, 256, 0, stream>>>(Wk3, wk3s, K2, K2, C_HID);     // 256 x 64
  edge_mlp_kernel<<<E_PAD / MTILE, EBLOCK, 0, stream>>>(
      pts, fyh, esrc, edst, wk1s, bk1, wk2s, bk2, wk3s, bk3, agg, cnt);
  final_mlp_kernel<<<BN, 64, 0, stream>>>(agg, cnt, Wp1, bp1, Wp2, bp2, (float*)d_out);
}
